// SingleAttention_7370163880734
// MI455X (gfx1250) — matrix operand order repaired and checked by reading, not compiled, not run on hardware
//
#include <hip/hip_runtime.h>

typedef __attribute__((ext_vector_type(16))) _Float16 v16h;
typedef __attribute__((ext_vector_type(8)))  float    v8f;
typedef int v4i __attribute__((vector_size(16)));

union Frag16 {
  v16h h;
  uint4 q[2];
  _Float16 e[16];
};

union Half8 {
  uint4 q;
  _Float16 e[8];
};

#define NB   8
#define SEQ  2048
#define DIN  512
#define DATT 128

#define WT_HALFS  (3 * DATT * DIN)     // 196608 halves
#define QKV_HALFS (NB * SEQ * DATT)    // 2097152 halves each

// ---- CDNA5 async global->LDS path (probe at compile time) -----------------
#if defined(__has_builtin)
#  if __has_builtin(__builtin_amdgcn_global_load_async_to_lds_b128)
#    define HAVE_ASYNC 1
#  endif
#  if __has_builtin(__builtin_amdgcn_s_wait_asynccnt)
#    define HAVE_WAITA 1
#  endif
#endif
#ifndef HAVE_ASYNC
#  define HAVE_ASYNC 0
#endif
#ifndef HAVE_WAITA
#  define HAVE_WAITA 0
#endif

typedef __attribute__((address_space(1))) v4i g_v4i;   // prints as "__device__"
typedef __attribute__((address_space(3))) v4i l_v4i;   // prints as "__shared__"

static __device__ __forceinline__ void async_cp16(const _Float16* g, _Float16* l) {
#if HAVE_ASYNC
  __builtin_amdgcn_global_load_async_to_lds_b128((g_v4i*)g, (l_v4i*)l, 0, 0);
#else
  *(uint4*)l = *(const uint4*)g;   // fallback: register relay
#endif
}

static __device__ __forceinline__ void wait_async_le4() {
#if HAVE_ASYNC
#  if HAVE_WAITA
  __builtin_amdgcn_s_wait_asynccnt(4);
#  else
  asm volatile("s_wait_asynccnt 0x4" ::: "memory");
#  endif
#endif
}

static __device__ __forceinline__ void wait_async_0() {
#if HAVE_ASYNC
#  if HAVE_WAITA
  __builtin_amdgcn_s_wait_asynccnt(0);
#  else
  asm volatile("s_wait_asynccnt 0x0" ::: "memory");
#  endif
#endif
}

static __device__ __forceinline__ v8f zero8() {
  v8f z = {0.f, 0.f, 0.f, 0.f, 0.f, 0.f, 0.f, 0.f};
  return z;
}

static __device__ __forceinline__ v8f wmma_f16(const Frag16& a, const Frag16& b, v8f c) {
  return __builtin_amdgcn_wmma_f32_16x16x32_f16(false, a.h, false, b.h, (short)0, c,
                                                false, false);
}

// ---------------------------------------------------------------------------
// Kernel 0: W[512][128] fp32 -> Wt[proj][128][512] f16 (transposed)
// ---------------------------------------------------------------------------
__global__ __launch_bounds__(256) void wprep_kernel(const float* __restrict__ Wq,
                                                    const float* __restrict__ Wk,
                                                    const float* __restrict__ Wv,
                                                    _Float16* __restrict__ wt) {
  int idx = blockIdx.x * 256 + threadIdx.x;
  if (idx >= 3 * DIN * DATT) return;
  int proj = idx >> 16;            // DIN*DATT == 65536
  int r    = idx & 65535;
  int d    = r >> 7;
  int f    = r & 127;
  const float* W = (proj == 0) ? Wq : (proj == 1 ? Wk : Wv);
  wt[(size_t)proj * DATT * DIN + (size_t)f * DIN + d] = (_Float16)W[d * DATT + f];
}

// ---------------------------------------------------------------------------
// Kernel 1: projections. One wave = 16 rows x 128 cols, K=512 in 16 steps.
// All 8 B-fragments preloaded per K-step so loads clause ahead of WMMAs.
// ---------------------------------------------------------------------------
__global__ __launch_bounds__(256) void proj_kernel(
    const float* __restrict__ qin, const float* __restrict__ kin,
    const float* __restrict__ vin, const _Float16* __restrict__ wt,
    const float* __restrict__ bq, const float* __restrict__ bk,
    const float* __restrict__ bv, _Float16* __restrict__ qh,
    _Float16* __restrict__ kh, _Float16* __restrict__ vth) {
  const int lane = threadIdx.x & 31;
  const int wave = threadIdx.x >> 5;
  const int l16  = lane & 15;
  const int lh   = lane >> 4;
  const int proj = blockIdx.x >> 7;
  const int row0 = (blockIdx.x & 127) * 128 + wave * 16;

  const float* in   = (proj == 0) ? qin : (proj == 1 ? kin : vin);
  const float* bias = (proj == 0) ? bq  : (proj == 1 ? bk  : bv);
  const _Float16* w = wt + (size_t)proj * DATT * DIN;

  v8f acc[8];
#pragma unroll
  for (int t = 0; t < 8; ++t) acc[t] = zero8();

  for (int ks = 0; ks < DIN; ks += 32) {
    const float* ap = in + (size_t)(row0 + l16) * DIN + ks + lh * 8;
    float4 f0 = *(const float4*)(ap);
    float4 f1 = *(const float4*)(ap + 4);
    float4 f2 = *(const float4*)(ap + 16);
    float4 f3 = *(const float4*)(ap + 20);

    // preload all 8 B fragments (W tiles), then fire WMMAs
    Frag16 bf[8];
#pragma unroll
    for (int nt = 0; nt < 8; ++nt) {
      const _Float16* wp = w + (size_t)(nt * 16 + l16) * DIN + ks + lh * 8;
      bf[nt].q[0] = *(const uint4*)(wp);
      bf[nt].q[1] = *(const uint4*)(wp + 16);
    }

    Frag16 a;
    a.e[0]  = (_Float16)f0.x; a.e[1]  = (_Float16)f0.y;
    a.e[2]  = (_Float16)f0.z; a.e[3]  = (_Float16)f0.w;
    a.e[4]  = (_Float16)f1.x; a.e[5]  = (_Float16)f1.y;
    a.e[6]  = (_Float16)f1.z; a.e[7]  = (_Float16)f1.w;
    a.e[8]  = (_Float16)f2.x; a.e[9]  = (_Float16)f2.y;
    a.e[10] = (_Float16)f2.z; a.e[11] = (_Float16)f2.w;
    a.e[12] = (_Float16)f3.x; a.e[13] = (_Float16)f3.y;
    a.e[14] = (_Float16)f3.z; a.e[15] = (_Float16)f3.w;

#pragma unroll
    for (int nt = 0; nt < 8; ++nt) acc[nt] = wmma_f16(a, bf[nt], acc[nt]);
  }

  if (proj < 2) {
    _Float16* outp = (proj == 0) ? qh : kh;
#pragma unroll
    for (int nt = 0; nt < 8; ++nt) {
      float bb = bias[nt * 16 + l16];
#pragma unroll
      for (int r = 0; r < 8; ++r)
        outp[(size_t)(row0 + r + lh * 8) * DATT + nt * 16 + l16] =
            (_Float16)(acc[nt][r] + bb);
    }
  } else {
    int bidx = row0 >> 11;
    int s0   = row0 & 2047;
#pragma unroll
    for (int nt = 0; nt < 8; ++nt) {
      float bb = bias[nt * 16 + l16];
      Half8 p;
#pragma unroll
      for (int r = 0; r < 8; ++r) p.e[r] = (_Float16)(acc[nt][r] + bb);
      *(uint4*)(vth + ((size_t)bidx * DATT + nt * 16 + l16) * SEQ + s0 + lh * 8) = p.q;
    }
  }
}

// ---------------------------------------------------------------------------
// Kernel 2: fused flash attention with double-buffered async global->LDS
// staging of K/V chunks (32 keys/iter). One wave = 16 queries x 128 dims.
// ---------------------------------------------------------------------------
__global__ __launch_bounds__(256) void attn_kernel(
    const _Float16* __restrict__ qh, const _Float16* __restrict__ kh,
    const _Float16* __restrict__ vth, float* __restrict__ out) {
  __shared__ _Float16 kbuf[2][32 * 128];   // 8 KB x2 : K chunk, key-major
  __shared__ _Float16 vbuf[2][128 * 32];   // 8 KB x2 : V chunk, dim-major
  __shared__ _Float16 pbuf[8][16 * 32];    // 1 KB per wave: P transpose tile

  const int tid  = threadIdx.x;
  const int lane = tid & 31;
  const int wave = tid >> 5;
  const int l16  = lane & 15;
  const int lh   = lane >> 4;
  const int b    = blockIdx.x >> 4;
  const int q0   = (blockIdx.x & 15) * 128 + wave * 16;

  const _Float16* kbase  = kh  + (size_t)b * SEQ * DATT;   // [key][128]
  const _Float16* vtbase = vth + (size_t)b * DATT * SEQ;   // [dim][SEQ]

  // ---- stage helper: 4 x b128 async copies per thread per chunk ----
  auto stage = [&](int buf, int kb) {
    const _Float16* kg = kbase + (size_t)kb * DATT;
#pragma unroll
    for (int r = 0; r < 2; ++r) {
      int idx = r * 256 + tid;                 // 0..511, 8 halves each
      async_cp16(kg + idx * 8, &kbuf[buf][idx * 8]);
    }
#pragma unroll
    for (int r = 0; r < 2; ++r) {
      int idx = r * 256 + tid;                 // 0..511
      int dim = idx >> 2, part = idx & 3;
      async_cp16(vtbase + (size_t)dim * SEQ + kb + part * 8, &vbuf[buf][idx * 8]);
    }
  };

  // Q A-fragments (registers, read once)
  Frag16 qf[4];
  const _Float16* qrow = qh + ((size_t)b * SEQ + q0 + l16) * DATT;
#pragma unroll
  for (int c = 0; c < 4; ++c) {
    qf[c].q[0] = *(const uint4*)(qrow + c * 32 + lh * 8);
    qf[c].q[1] = *(const uint4*)(qrow + c * 32 + lh * 8 + 16);
  }

  v8f o[8];
#pragma unroll
  for (int t = 0; t < 8; ++t) o[t] = zero8();
  float mrow[8], lrow[8];
#pragma unroll
  for (int r = 0; r < 8; ++r) { mrow[r] = -1e30f; lrow[r] = 0.f; }

  const float sc = 0.08838834764831845f * 1.4426950408889634f;  // scale*log2e

  stage(0, 0);   // prologue DMA

  const int NIT = SEQ / 32;
  for (int it = 0; it < NIT; ++it) {
    const int cur = it & 1;
    if (it + 1 < NIT) {
      stage(cur ^ 1, (it + 1) * 32);   // overlap next chunk's DMA
      wait_async_le4();                // in-order: current chunk complete
    } else {
      wait_async_0();
    }
    __syncthreads();                   // all waves' staging visible

    // ---- scores: 8 WMMAs out of LDS ----
    v8f s0 = zero8(), s1 = zero8();
    const _Float16* krow0 = &kbuf[cur][(size_t)l16 * DATT + lh * 8];
    const _Float16* krow1 = krow0 + 16 * DATT;
#pragma unroll
    for (int c = 0; c < 4; ++c) {
      Frag16 b0, b1;
      b0.q[0] = *(const uint4*)(krow0 + c * 32);
      b0.q[1] = *(const uint4*)(krow0 + c * 32 + 16);
      b1.q[0] = *(const uint4*)(krow1 + c * 32);
      b1.q[1] = *(const uint4*)(krow1 + c * 32 + 16);
      s0 = wmma_f16(qf[c], b0, s0);
      s1 = wmma_f16(qf[c], b1, s1);
    }

    // ---- online softmax (row r lives in VGPR r across a 16-lane half) ----
    float alpha[8];
#pragma unroll
    for (int r = 0; r < 8; ++r) {
      float t0 = s0[r] * sc, t1 = s1[r] * sc;
      float mx = fmaxf(t0, t1);
#pragma unroll
      for (int off = 1; off < 16; off <<= 1)
        mx = fmaxf(mx, __shfl_xor(mx, off, 32));
      float mnew = fmaxf(mrow[r], mx);
      float p0 = exp2f(t0 - mnew);
      float p1 = exp2f(t1 - mnew);
      float rs = p0 + p1;
#pragma unroll
      for (int off = 1; off < 16; off <<= 1)
        rs += __shfl_xor(rs, off, 32);
      alpha[r] = exp2f(mrow[r] - mnew);
      lrow[r]  = lrow[r] * alpha[r] + rs;
      mrow[r]  = mnew;
      pbuf[wave][(r + lh * 8) * 32 + l16]      = (_Float16)p0;
      pbuf[wave][(r + lh * 8) * 32 + l16 + 16] = (_Float16)p1;
    }
#pragma unroll
    for (int t = 0; t < 8; ++t)
#pragma unroll
      for (int r = 0; r < 8; ++r) o[t][r] *= alpha[r];

    __syncthreads();                   // P store -> P load ordering

    // ---- P x V: 8 WMMAs out of LDS ----
    Frag16 pf;
    const _Float16* pr = &pbuf[wave][l16 * 32 + lh * 8];
    pf.q[0] = *(const uint4*)(pr);
    pf.q[1] = *(const uint4*)(pr + 16);

    const _Float16* vb = &vbuf[cur][lh * 8];
#pragma unroll
    for (int t = 0; t < 8; ++t) {
      const _Float16* vr = vb + (size_t)(t * 16 + l16) * 32;
      Frag16 vf;
      vf.q[0] = *(const uint4*)vr;
      vf.q[1] = *(const uint4*)(vr + 16);
      o[t] = wmma_f16(pf, vf, o[t]);
    }
    __syncthreads();                   // readers done before buffer reuse
  }

  float* orow = out + ((size_t)b * SEQ + q0) * DATT;
  float inv[8];
#pragma unroll
  for (int r = 0; r < 8; ++r) inv[r] = 1.0f / lrow[r];
#pragma unroll
  for (int t = 0; t < 8; ++t)
#pragma unroll
    for (int r = 0; r < 8; ++r)
      orow[(size_t)(r + lh * 8) * DATT + t * 16 + l16] = o[t][r] * inv[r];
}

// ---------------------------------------------------------------------------
extern "C" void kernel_launch(void* const* d_in, const int* in_sizes, int n_in,
                              void* d_out, int out_size, void* d_ws, size_t ws_size,
                              hipStream_t stream) {
  (void)in_sizes; (void)n_in; (void)out_size; (void)ws_size;
  const float* qin = (const float*)d_in[0];
  const float* kin = (const float*)d_in[1];
  const float* vin = (const float*)d_in[2];
  const float* Wq  = (const float*)d_in[3];
  const float* Wk  = (const float*)d_in[4];
  const float* Wv  = (const float*)d_in[5];
  const float* bq  = (const float*)d_in[6];
  const float* bk  = (const float*)d_in[7];
  const float* bv  = (const float*)d_in[8];

  _Float16* wt  = (_Float16*)d_ws;
  _Float16* qhp = wt + WT_HALFS;
  _Float16* khp = qhp + QKV_HALFS;
  _Float16* vtp = khp + QKV_HALFS;

  wprep_kernel<<<(3 * DIN * DATT + 255) / 256, 256, 0, stream>>>(Wq, Wk, Wv, wt);
  proj_kernel<<<3 * 128, 256, 0, stream>>>(qin, kin, vin, wt, bq, bk, bv,
                                           qhp, khp, vtp);
  attn_kernel<<<NB * 16, 256, 0, stream>>>(qhp, khp, vtp, (float*)d_out);
}
